// CC_AlexNet_56710748176568
// MI455X (gfx1250) — matrix operand order repaired and checked by reading, not compiled, not run on hardware
//
#include <hip/hip_runtime.h>
#include <hip/hip_bf16.h>

typedef __attribute__((ext_vector_type(16))) __bf16 v16bf;
typedef __attribute__((ext_vector_type(8)))  float  v8f;

#define NEXP 8
#define BATCH 1024

// ---------------------------------------------------------------------------
// fp32 -> bf16 bulk convert (for the network input x)
// ---------------------------------------------------------------------------
__global__ void f32_to_bf16(const float* __restrict__ in, __bf16* __restrict__ out,
                            long n) {
  for (long i = (long)blockIdx.x * blockDim.x + threadIdx.x; i < n;
       i += (long)gridDim.x * blockDim.x)
    out[i] = (__bf16)in[i];
}

// ---------------------------------------------------------------------------
// Weight repack: fp32 [E][M][Cin][3][3] (or fp32 [M][K] for FC) -> bf16 in
// WMMA A-fragment order: idx = (((e*Mtiles+mt)*Ktiles+kt)*32 + lane)*16 + j
// 16-bit A-matrix 16x32, contiguous k order used here (B matches): lane L: m = L%16,
// element j: k = (L/16)*16 + j
// ---------------------------------------------------------------------------
__global__ void repack_weights(const float* __restrict__ w, __bf16* __restrict__ out,
                               int nexp, int M, int Kreal, int Mtiles, int Ktiles,
                               int is_conv) {
  size_t total = (size_t)nexp * Mtiles * Ktiles * 512;
  for (size_t idx = (size_t)blockIdx.x * blockDim.x + threadIdx.x; idx < total;
       idx += (size_t)gridDim.x * blockDim.x) {
    int j = (int)(idx & 15);
    int L = (int)((idx >> 4) & 31);
    size_t t = idx >> 9;
    int kt = (int)(t % (size_t)Ktiles); t /= (size_t)Ktiles;
    int mt = (int)(t % (size_t)Mtiles);
    int e  = (int)(t / (size_t)Mtiles);
    int m = mt * 16 + (L & 15);
    int klocal = ((L >> 4) << 4) + j;
    int k = kt * 32 + klocal;
    float v = 0.f;
    if (m < M && k < Kreal) {
      if (is_conv) {
        int Cin = Kreal / 9;
        int cin = k / 9, rem = k % 9, ky = rem / 3, kx = rem % 3;
        v = w[((((size_t)e * M + m) * Cin + cin) * 3 + ky) * 3 + kx];
      } else {
        v = w[(size_t)m * Kreal + k];
      }
    }
    out[idx] = (__bf16)v;
  }
}

// ---------------------------------------------------------------------------
// Routing: pooled mean over HxW (bf16 input), r = sigmoid(pooled @ rw^T + rb)
// ---------------------------------------------------------------------------
__global__ __launch_bounds__(256) void routing_kernel(const __bf16* __restrict__ in,
                                                      const float* __restrict__ rw,
                                                      const float* __restrict__ rb,
                                                      float* __restrict__ r,
                                                      int Cin, int HW) {
  __shared__ float pooled[384];
  int b = blockIdx.x;
  const __bf16* inb = in + (size_t)b * Cin * HW;
  for (int c = threadIdx.x; c < Cin; c += blockDim.x) {
    float s = 0.f;
    const __bf16* p = inb + (size_t)c * HW;
    for (int i = 0; i < HW; ++i) s += (float)p[i];
    pooled[c] = s / (float)HW;
  }
  __syncthreads();
  if (threadIdx.x < NEXP) {
    int e = threadIdx.x;
    float l = rb[e];
    for (int c = 0; c < Cin; ++c) l += pooled[c] * rw[e * Cin + c];
    r[b * NEXP + e] = 1.f / (1.f + __expf(-l));
  }
}

// ---------------------------------------------------------------------------
// CondConv as WMMA implicit GEMM, expert sum folded in epilogue:
//   y[b,:,:] = ReLU(BN( sum_e r_be * (W_e @ im2col(x_b)) + bias ))
// Workgroup = (example b, 16-pixel N-tile). 4 waves partition M-tiles.
// Inner loop: one B fragment from LDS feeds 8 independent expert WMMA chains
// (8x less LDS traffic, 8-deep WMMA ILP for the XDL pipeline).
// ---------------------------------------------------------------------------
struct ConvParams {
  const __bf16* in;         // [B, Cin, Hin, Win] bf16
  const __bf16* w;          // repacked expert weights
  const float* r;           // [B, 8] routing
  const float* eb;          // [8, Cout] expert bias
  const float* gamma; const float* beta; const float* mean; const float* var;
  __bf16* out;              // [B, Cout, Hout, Wout] bf16
  int Cin, Hin, Win, Cout, Hout, Wout, stride;
  int K, Kreal, Mtiles, Ktiles, mt_per_wave;
};

__global__ __launch_bounds__(128) void condconv_wmma(ConvParams p) {
  extern __shared__ __bf16 lds[];             // im2col tile: [16 pixels][K] bf16
  const int b    = blockIdx.x;
  const int nt   = blockIdx.y;
  const int tid  = threadIdx.x;
  const int lane = tid & 31;
  const int wave = tid >> 5;

  const int HW      = p.Hout * p.Wout;
  const int pixbase = nt * 16;
  const __bf16* inb = p.in + (size_t)b * p.Cin * p.Hin * p.Win;

  // ---- stage im2col (bf16), zero-pad K tail and spatial halo
  for (int idx = tid; idx < 16 * p.K; idx += 128) {
    int n = idx / p.K;
    int k = idx - n * p.K;
    __bf16 v = (__bf16)0.f;
    if (k < p.Kreal) {
      int pix = pixbase + n;
      int oy = pix / p.Wout, ox = pix - oy * p.Wout;
      int cin = k / 9, rem = k - cin * 9;
      int ky = rem / 3, kx = rem - ky * 3;
      int iy = oy * p.stride + ky - 1;
      int ix = ox * p.stride + kx - 1;
      if (iy >= 0 && iy < p.Hin && ix >= 0 && ix < p.Win)
        v = inb[((size_t)cin * p.Hin + iy) * p.Win + ix];
    }
    lds[idx] = v;
  }
  __syncthreads();

  float rmix[NEXP];
  #pragma unroll
  for (int e = 0; e < NEXP; ++e) rmix[e] = p.r[b * NEXP + e];

  const int nlane = lane & 15;
  const int khalf = lane >> 4;
  const __bf16* ldsb = lds + (size_t)nlane * p.K + khalf * 16;
  const size_t estride = (size_t)p.Mtiles * p.Ktiles * 512;

  for (int mi = 0; mi < p.mt_per_wave; ++mi) {
    int mt = wave * p.mt_per_wave + mi;     // wave-uniform
    if (mt >= p.Mtiles) break;

    v8f acc[NEXP];
    #pragma unroll
    for (int e = 0; e < NEXP; ++e)
      #pragma unroll
      for (int i = 0; i < 8; ++i) acc[e][i] = 0.f;

    const __bf16* wp = p.w + ((size_t)mt * p.Ktiles) * 512 + lane * 16; // expert 0
    #pragma unroll 2
    for (int kt = 0; kt < p.Ktiles; ++kt) {
      v16bf bfrag = *(const v16bf*)(ldsb + kt * 32);
      __builtin_prefetch(wp + 512, 0, 1);
      #pragma unroll
      for (int e = 0; e < NEXP; ++e) {
        v16bf a = *(const v16bf*)(wp + (size_t)e * estride);
        acc[e] = __builtin_amdgcn_wmma_f32_16x16x32_bf16(
            false, a, false, bfrag, (short)0, acc[e], false, false);
      }
      wp += 512;
    }

    // ---- combine experts with routing weights
    v8f accT;
    #pragma unroll
    for (int i = 0; i < 8; ++i) accT[i] = 0.f;
    #pragma unroll
    for (int e = 0; e < NEXP; ++e) {
      float re = rmix[e];
      #pragma unroll
      for (int i = 0; i < 8; ++i) accT[i] += re * acc[e][i];
    }

    // ---- epilogue: routing bias + BN + ReLU; C layout: M = i + 8*khalf, N = nlane
    #pragma unroll
    for (int i = 0; i < 8; ++i) {
      int m = mt * 16 + i + khalf * 8;
      float bias = 0.f;
      #pragma unroll
      for (int e = 0; e < NEXP; ++e) bias += rmix[e] * p.eb[e * p.Cout + m];
      float inv = rsqrtf(p.var[m] + 1e-5f);
      float sc  = p.gamma[m] * inv;
      float sh  = p.beta[m] - p.mean[m] * sc;
      float v = (accT[i] + bias) * sc + sh;
      v = v > 0.f ? v : 0.f;
      p.out[((size_t)b * p.Cout + m) * HW + pixbase + nlane] = (__bf16)v;
    }
  }
}

// ---------------------------------------------------------------------------
// 2x2 max pool, stride 2 (bf16)
// ---------------------------------------------------------------------------
__global__ void maxpool_kernel(const __bf16* __restrict__ in, __bf16* __restrict__ out,
                               int C, int H, int W, long total) {
  int Wo = W >> 1, Ho = H >> 1;
  for (long idx = (long)blockIdx.x * blockDim.x + threadIdx.x; idx < total;
       idx += (long)gridDim.x * blockDim.x) {
    long t = idx;
    int x = (int)(t % Wo); t /= Wo;
    int y = (int)(t % Ho); t /= Ho;
    int c = (int)(t % C);
    int b = (int)(t / C);
    const __bf16* p = in + (((size_t)b * C + c) * H + y * 2) * W + x * 2;
    float m = fmaxf(fmaxf((float)p[0], (float)p[1]),
                    fmaxf((float)p[W], (float)p[W + 1]));
    out[idx] = (__bf16)m;
  }
}

// ---------------------------------------------------------------------------
// FC as WMMA GEMM. B-tile = 16 consecutive examples' activations, which is
// one contiguous 32*K-byte block -> staged with async global->LDS copies
// (GLOBAL_LOAD_ASYNC_TO_LDS_B128, tracked by ASYNCcnt).
// ---------------------------------------------------------------------------
struct FcParams {
  const __bf16* in;         // [B, K] bf16
  const __bf16* w;          // repacked
  const float* bias;        // [Mout]
  void* out;                // [B, Mout] (bf16 or fp32)
  int K, Ktiles, Mtiles, Mout, mt_per_wave, relu, out_f32;
};

__global__ __launch_bounds__(128) void fc_wmma(FcParams p) {
  extern __shared__ __bf16 ldsf[];            // [16 examples][K] bf16
  const int btile = blockIdx.x;
  const int tid = threadIdx.x, lane = tid & 31, wave = tid >> 5;

  // ---- async bulk copy: 32*K bytes contiguous, 16B chunks per lane
  {
    unsigned ldsbase = (unsigned)(size_t)(void*)ldsf;   // LDS byte address
    const __bf16* src = p.in + (size_t)btile * 16 * p.K;
    int nchunk = 2 * p.K;                               // (16*K*2)/16
    for (int i = tid; i < nchunk; i += 128) {
      unsigned dst = ldsbase + (unsigned)i * 16u;
      unsigned long long ga = (unsigned long long)(size_t)(src + (size_t)i * 8);
      asm volatile("global_load_async_to_lds_b128 %0, %1, off"
                   :: "v"(dst), "v"(ga) : "memory");
    }
    asm volatile("s_wait_asynccnt 0x0" ::: "memory");
  }
  __syncthreads();

  const int nlane = lane & 15;
  const int khalf = lane >> 4;
  const __bf16* ldsb = ldsf + (size_t)nlane * p.K + khalf * 16;

  for (int mi = 0; mi < p.mt_per_wave; ++mi) {
    int mt = blockIdx.y * (4 * p.mt_per_wave) + wave * p.mt_per_wave + mi;
    if (mt >= p.Mtiles) break;                // wave-uniform
    v8f acc;
    #pragma unroll
    for (int i = 0; i < 8; ++i) acc[i] = 0.f;
    const __bf16* wp = p.w + ((size_t)mt * p.Ktiles) * 512 + lane * 16;
    #pragma unroll 4
    for (int kt = 0; kt < p.Ktiles; ++kt) {
      __builtin_prefetch(wp + 512, 0, 1);
      v16bf a  = *(const v16bf*)wp;  wp += 512;
      v16bf bf = *(const v16bf*)(ldsb + kt * 32);
      acc = __builtin_amdgcn_wmma_f32_16x16x32_bf16(
          false, a, false, bf, (short)0, acc, false, false);
    }
    #pragma unroll
    for (int i = 0; i < 8; ++i) {
      int m = mt * 16 + i + khalf * 8;
      if (m < p.Mout) {
        float v = acc[i] + p.bias[m];
        if (p.relu) v = v > 0.f ? v : 0.f;
        size_t o = (size_t)(btile * 16 + nlane) * p.Mout + m;
        if (p.out_f32) ((float*)p.out)[o] = v;
        else           ((__bf16*)p.out)[o] = (__bf16)v;
      }
    }
  }
}

// ---------------------------------------------------------------------------
// Host orchestration
// ---------------------------------------------------------------------------
struct LayerCfg {
  int Cin, Cout, Hin, Win, Hout, Wout, stride;
  int K, Kreal, Mtiles, Ktiles, mtpw, ntiles;
};

static const LayerCfg LC[5] = {
  {  3,  64, 32, 32, 16, 16, 2,   32,   27,  4,   1, 1, 16},
  { 64, 192,  8,  8,  8,  8, 1,  576,  576, 12,  18, 3,  4},
  {192, 384,  4,  4,  4,  4, 1, 1728, 1728, 24,  54, 6,  1},
  {384, 256,  4,  4,  4,  4, 1, 3456, 3456, 16, 108, 4,  1},
  {256, 256,  4,  4,  4,  4, 1, 2304, 2304, 16,  72, 4,  1},
};

extern "C" void kernel_launch(void* const* d_in, const int* in_sizes, int n_in,
                              void* d_out, int out_size, void* d_ws, size_t ws_size,
                              hipStream_t stream) {
  (void)in_sizes; (void)n_in; (void)out_size;
  // ---- input unpack (setup_inputs insertion order)
  const float* x = (const float*)d_in[0];
  const float *c_rw[5], *c_rb[5], *c_w[5], *c_b[5];
  for (int l = 0; l < 5; ++l) {
    c_rw[l] = (const float*)d_in[1 + l * 4 + 0];
    c_rb[l] = (const float*)d_in[1 + l * 4 + 1];
    c_w[l]  = (const float*)d_in[1 + l * 4 + 2];
    c_b[l]  = (const float*)d_in[1 + l * 4 + 3];
  }
  const float *bn_g[5], *bn_b[5], *bn_m[5], *bn_v[5];
  for (int l = 0; l < 5; ++l) {
    bn_g[l] = (const float*)d_in[21 + l * 4 + 0];
    bn_b[l] = (const float*)d_in[21 + l * 4 + 1];
    bn_m[l] = (const float*)d_in[21 + l * 4 + 2];
    bn_v[l] = (const float*)d_in[21 + l * 4 + 3];
  }
  const float *fc_w[3], *fc_b[3];
  for (int f = 0; f < 3; ++f) {
    fc_w[f] = (const float*)d_in[41 + f * 2 + 0];
    fc_b[f] = (const float*)d_in[41 + f * 2 + 1];
  }

  // ---- workspace bump allocator
  char* ws = (char*)d_ws;
  size_t off = 0;
  auto alloc = [&](size_t bytes) -> void* {
    void* p = ws + off;
    off = (off + bytes + 255) & ~(size_t)255;
    return p;
  };

  __bf16* wpk[5];
  for (int l = 0; l < 5; ++l)
    wpk[l] = (__bf16*)alloc((size_t)NEXP * LC[l].Mtiles * LC[l].Ktiles * 512 * 2);
  static const int fcK[3]   = {1024, 4096, 4096};
  static const int fcM[3]   = {4096, 4096, 100};
  static const int fcMt[3]  = {256, 256, 7};
  static const int fcKt[3]  = {32, 128, 128};
  __bf16* fwpk[3];
  for (int f = 0; f < 3; ++f)
    fwpk[f] = (__bf16*)alloc((size_t)fcMt[f] * fcKt[f] * 512 * 2);

  float* rbuf[5];
  for (int l = 0; l < 5; ++l) rbuf[l] = (float*)alloc((size_t)BATCH * NEXP * 4);

  __bf16* xbf = (__bf16*)alloc((size_t)BATCH * 3 * 32 * 32 * 2);
  __bf16* a0 = (__bf16*)alloc((size_t)BATCH * 64  * 16 * 16 * 2);
  __bf16* p0 = (__bf16*)alloc((size_t)BATCH * 64  *  8 *  8 * 2);
  __bf16* a1 = (__bf16*)alloc((size_t)BATCH * 192 *  8 *  8 * 2);
  __bf16* p1 = (__bf16*)alloc((size_t)BATCH * 192 *  4 *  4 * 2);
  __bf16* a2 = (__bf16*)alloc((size_t)BATCH * 384 *  4 *  4 * 2);
  __bf16* a3 = (__bf16*)alloc((size_t)BATCH * 256 *  4 *  4 * 2);
  __bf16* a4 = (__bf16*)alloc((size_t)BATCH * 256 *  4 *  4 * 2);
  __bf16* p4 = (__bf16*)alloc((size_t)BATCH * 1024 * 2);        // flattened FC input
  __bf16* f0 = (__bf16*)alloc((size_t)BATCH * 4096 * 2);
  __bf16* f1 = (__bf16*)alloc((size_t)BATCH * 4096 * 2);
  if (off > ws_size) { /* workspace too small; nothing safe to do */ }

  // ---- convert x, repack all weights (deterministic, every call)
  f32_to_bf16<<<dim3(2048), dim3(256), 0, stream>>>(
      x, xbf, (long)BATCH * 3 * 32 * 32);
  for (int l = 0; l < 5; ++l)
    repack_weights<<<dim3(1024), dim3(256), 0, stream>>>(
        c_w[l], wpk[l], NEXP, LC[l].Cout, LC[l].Kreal, LC[l].Mtiles, LC[l].Ktiles, 1);
  for (int f = 0; f < 3; ++f)
    repack_weights<<<dim3(1024), dim3(256), 0, stream>>>(
        fc_w[f], fwpk[f], 1, fcM[f], fcK[f], fcMt[f], fcKt[f], 0);

  // ---- conv stack
  const __bf16* conv_in[5]  = {xbf, p0, p1, a2, a3};
  __bf16*       conv_out[5] = {a0, a1, a2, a3, a4};
  const int     rHW[5]      = {1024, 64, 16, 16, 16};

  for (int l = 0; l < 5; ++l) {
    routing_kernel<<<dim3(BATCH), dim3(256), 0, stream>>>(
        conv_in[l], c_rw[l], c_rb[l], rbuf[l], LC[l].Cin, rHW[l]);

    ConvParams cp;
    cp.in = conv_in[l]; cp.w = wpk[l]; cp.r = rbuf[l]; cp.eb = c_b[l];
    cp.gamma = bn_g[l]; cp.beta = bn_b[l]; cp.mean = bn_m[l]; cp.var = bn_v[l];
    cp.out = conv_out[l];
    cp.Cin = LC[l].Cin; cp.Hin = LC[l].Hin; cp.Win = LC[l].Win;
    cp.Cout = LC[l].Cout; cp.Hout = LC[l].Hout; cp.Wout = LC[l].Wout;
    cp.stride = LC[l].stride; cp.K = LC[l].K; cp.Kreal = LC[l].Kreal;
    cp.Mtiles = LC[l].Mtiles; cp.Ktiles = LC[l].Ktiles; cp.mt_per_wave = LC[l].mtpw;
    size_t shmem = (size_t)16 * LC[l].K * 2;
    condconv_wmma<<<dim3(BATCH, LC[l].ntiles), dim3(128), shmem, stream>>>(cp);

    if (l == 0) {
      long total = (long)BATCH * 64 * 8 * 8;
      maxpool_kernel<<<dim3(2048), dim3(256), 0, stream>>>(a0, p0, 64, 16, 16, total);
    } else if (l == 1) {
      long total = (long)BATCH * 192 * 4 * 4;
      maxpool_kernel<<<dim3(2048), dim3(256), 0, stream>>>(a1, p1, 192, 8, 8, total);
    } else if (l == 4) {
      long total = (long)BATCH * 256 * 2 * 2;
      maxpool_kernel<<<dim3(1024), dim3(256), 0, stream>>>(a4, p4, 256, 4, 4, total);
    }
  }

  // ---- FC stack (p4 is the NCHW-contiguous flatten [B,1024] in bf16)
  {
    FcParams fp;
    fp.in = p4; fp.w = fwpk[0]; fp.bias = fc_b[0]; fp.out = f0;
    fp.K = 1024; fp.Ktiles = 32; fp.Mtiles = 256; fp.Mout = 4096;
    fp.mt_per_wave = 4; fp.relu = 1; fp.out_f32 = 0;
    fc_wmma<<<dim3(BATCH / 16, 16), dim3(128), (size_t)16 * 1024 * 2, stream>>>(fp);
  }
  {
    FcParams fp;
    fp.in = f0; fp.w = fwpk[1]; fp.bias = fc_b[1]; fp.out = f1;
    fp.K = 4096; fp.Ktiles = 128; fp.Mtiles = 256; fp.Mout = 4096;
    fp.mt_per_wave = 4; fp.relu = 1; fp.out_f32 = 0;
    fc_wmma<<<dim3(BATCH / 16, 16), dim3(128), (size_t)16 * 4096 * 2, stream>>>(fp);
  }
  {
    FcParams fp;
    fp.in = f1; fp.w = fwpk[2]; fp.bias = fc_b[2]; fp.out = d_out;
    fp.K = 4096; fp.Ktiles = 128; fp.Mtiles = 7; fp.Mout = 100;
    fp.mt_per_wave = 2; fp.relu = 0; fp.out_f32 = 1;
    fc_wmma<<<dim3(BATCH / 16, 1), dim3(128), (size_t)16 * 4096 * 2, stream>>>(fp);
  }
}
